// QuantumResidualBlock_39049842655625
// MI455X (gfx1250) — compile-verified, matrix operand order checked by reading, not run on hardware
//
#include <hip/hip_runtime.h>
#include <math.h>

#define NQ 6
#define IN_DIM 512
#define OUT_DIM 512
#define ROWS 32768            // 8 * 4096
#define LDA 520               // bf16 elements per LDS A row (512 + 8 pad)

typedef __attribute__((ext_vector_type(16))) __bf16 v16bf;
typedef __attribute__((ext_vector_type(8)))  float  v8f;

struct c2 { float x, y; };
__device__ __forceinline__ c2 cmul(c2 a, c2 b) { return {a.x*b.x - a.y*b.y, a.x*b.y + a.y*b.x}; }
__device__ __forceinline__ c2 cadd(c2 a, c2 b) { return {a.x + b.x, a.y + b.y}; }

__device__ __forceinline__ unsigned short f2bf(float f) {
  unsigned u = __float_as_uint(f);
  unsigned r = ((u >> 16) & 1u) + 0x7FFFu;   // round to nearest even
  return (unsigned short)((u + r) >> 16);
}

// ---------- quantum gate helpers (1 row per wave32; lane holds amps s=lane, s=lane+32) ----------

__device__ __forceinline__ void apply1q(int q, int lane, c2 u00, c2 u01, c2 u10, c2 u11,
                                        c2& lo, c2& hi) {
  if (q == 5) {                               // bit5 lives in-register
    c2 nlo = cadd(cmul(u00, lo), cmul(u01, hi));
    c2 nhi = cadd(cmul(u10, lo), cmul(u11, hi));
    lo = nlo; hi = nhi;
  } else {
    const int m = 1 << q;
    c2 plo, phi;
    plo.x = __shfl_xor(lo.x, m, 32); plo.y = __shfl_xor(lo.y, m, 32);
    phi.x = __shfl_xor(hi.x, m, 32); phi.y = __shfl_xor(hi.y, m, 32);
    const bool b = (lane >> q) & 1;
    c2 ud = b ? u11 : u00;
    c2 uo = b ? u10 : u01;
    lo = cadd(cmul(ud, lo), cmul(uo, plo));
    hi = cadd(cmul(ud, hi), cmul(uo, phi));
  }
}

__device__ __forceinline__ void applyRZ(int q, int lane, float ch, float sh, c2& lo, c2& hi) {
  // diag(e^{-i t/2}, e^{+i t/2}); ch=cos(t/2), sh=sin(t/2)
  if (q == 5) {
    lo = cmul(lo, {ch, -sh});
    hi = cmul(hi, {ch,  sh});
  } else {
    float s = ((lane >> q) & 1) ? sh : -sh;
    c2 p = {ch, s};
    lo = cmul(lo, p);
    hi = cmul(hi, p);
  }
}

__device__ __forceinline__ void cnotg(int c, int t, int lane, c2& lo, c2& hi) {
  if (t == 5) {                               // ctrl < 5: in-register swap
    if ((lane >> c) & 1) { c2 tmp = lo; lo = hi; hi = tmp; }
  } else if (c == 5) {                        // only hi half (bit5=1) swaps
    const int m = 1 << t;
    c2 ph; ph.x = __shfl_xor(hi.x, m, 32); ph.y = __shfl_xor(hi.y, m, 32);
    hi = ph;
  } else {
    const int m = 1 << t;
    c2 pl, ph;
    pl.x = __shfl_xor(lo.x, m, 32); pl.y = __shfl_xor(lo.y, m, 32);
    ph.x = __shfl_xor(hi.x, m, 32); ph.y = __shfl_xor(hi.y, m, 32);
    if ((lane >> c) & 1) { lo = pl; hi = ph; }
  }
}

// ---------- kernel 1: skip_w f32 -> bf16 ----------

__global__ void cvt_wb_kernel(const float* __restrict__ w, unsigned short* __restrict__ bw, int n) {
  int i = blockIdx.x * blockDim.x + threadIdx.x;
  if (i < n) bw[i] = f2bf(w[i]);
}

// ---------- kernel 2: z = x@entry_w^T + b ; q = VQC(z) ; s = z + q ----------

__global__ __launch_bounds__(256) void vqc_kernel(
    const float* __restrict__ x, const float* __restrict__ entry_w,
    const float* __restrict__ entry_b, const float* __restrict__ vqc_w,
    float* __restrict__ sout) {
  __shared__ float ew[NQ * IN_DIM];
  const int tid = threadIdx.x;
  const int lane = tid & 31;
  const int wv = tid >> 5;
  for (int i = tid; i < NQ * IN_DIM; i += 256) ew[i] = entry_w[i];
  __syncthreads();

  const size_t row = (size_t)blockIdx.x * 8 + wv;
  const float* xr = x + row * IN_DIM + lane * 16;
  float4 a0 = *(const float4*)(xr + 0);
  float4 a1 = *(const float4*)(xr + 4);
  float4 a2 = *(const float4*)(xr + 8);
  float4 a3 = *(const float4*)(xr + 12);

  float z[NQ];
#pragma unroll
  for (int j = 0; j < NQ; ++j) {
    const float* wj = ew + j * IN_DIM + lane * 16;
    float4 b0 = *(const float4*)(wj + 0);
    float4 b1 = *(const float4*)(wj + 4);
    float4 b2 = *(const float4*)(wj + 8);
    float4 b3 = *(const float4*)(wj + 12);
    float d = a0.x*b0.x + a0.y*b0.y + a0.z*b0.z + a0.w*b0.w
            + a1.x*b1.x + a1.y*b1.y + a1.z*b1.z + a1.w*b1.w
            + a2.x*b2.x + a2.y*b2.y + a2.z*b2.z + a2.w*b2.w
            + a3.x*b3.x + a3.y*b3.y + a3.z*b3.z + a3.w*b3.w;
#pragma unroll
    for (int m = 16; m >= 1; m >>= 1) d += __shfl_xor(d, m, 32);
    z[j] = d + entry_b[j];
  }

  float ryc[NQ], rys[NQ], rzc[NQ], rzs[NQ];
#pragma unroll
  for (int j = 0; j < NQ; ++j) {
    float t = 0.5f * atanf(z[j]);
    ryc[j] = cosf(t); rys[j] = sinf(t);
    float t2 = 0.5f * atanf(z[j] * z[j]);
    rzc[j] = cosf(t2); rzs[j] = sinf(t2);
  }

  c2 lo = {(lane == 0) ? 1.f : 0.f, 0.f};
  c2 hi = {0.f, 0.f};
  const float R = 0.70710678118654752f;

  // encoding: H, RY(atan z), RZ(atan z^2) per qubit
#pragma unroll
  for (int i = 0; i < NQ; ++i) {
    apply1q(i, lane, {R, 0.f}, {R, 0.f}, {R, 0.f}, {-R, 0.f}, lo, hi);
    apply1q(i, lane, {ryc[i], 0.f}, {-rys[i], 0.f}, {rys[i], 0.f}, {ryc[i], 0.f}, lo, hi);
    applyRZ(i, lane, rzc[i], rzs[i], lo, hi);
  }

#pragma unroll
  for (int layer = 0; layer < 2; ++layer) {
#pragma unroll
    for (int i = 0; i < NQ; ++i) cnotg(i, (i + 1) % NQ, lane, lo, hi);
#pragma unroll
    for (int i = 0; i < NQ; ++i) {
      float wx = vqc_w[layer * 18 + 0 * 6 + i];
      float wy = vqc_w[layer * 18 + 1 * 6 + i];
      float wz = vqc_w[layer * 18 + 2 * 6 + i];
      float cx = cosf(0.5f * wx), sx = sinf(0.5f * wx);
      float cy = cosf(0.5f * wy), sy = sinf(0.5f * wy);
      float cz = cosf(0.5f * wz), sz = sinf(0.5f * wz);
      apply1q(i, lane, {cx, 0.f}, {0.f, -sx}, {0.f, -sx}, {cx, 0.f}, lo, hi);   // RX
      apply1q(i, lane, {cy, 0.f}, {-sy, 0.f}, {sy, 0.f}, {cy, 0.f}, lo, hi);    // RY
      applyRZ(i, lane, cz, sz, lo, hi);                                          // RZ
    }
    if (layer == 0) {  // data re-upload
#pragma unroll
      for (int i = 0; i < NQ; ++i) {
        apply1q(i, lane, {ryc[i], 0.f}, {-rys[i], 0.f}, {rys[i], 0.f}, {ryc[i], 0.f}, lo, hi);
        applyRZ(i, lane, rzc[i], rzs[i], lo, hi);
      }
    }
  }

  float pl = lo.x * lo.x + lo.y * lo.y;
  float ph = hi.x * hi.x + hi.y * hi.y;
  float e[NQ];
#pragma unroll
  for (int i = 0; i < 5; ++i) e[i] = (((lane >> i) & 1) ? -1.f : 1.f) * (pl + ph);
  e[5] = pl - ph;
#pragma unroll
  for (int i = 0; i < NQ; ++i) {
#pragma unroll
    for (int m = 16; m >= 1; m >>= 1) e[i] += __shfl_xor(e[i], m, 32);
  }
  if (lane == 0) {
#pragma unroll
    for (int j = 0; j < NQ; ++j) sout[row * NQ + j] = z[j] + e[j];
  }
}

// ---------- kernel 3: bf16 WMMA skip GEMM + exit GEMM + fused LayerNorm ----------

__device__ __forceinline__ v8f wmma_bf16(v16bf a, v16bf b, v8f c) {
  return __builtin_amdgcn_wmma_f32_16x16x32_bf16(false, a, false, b, (short)0, c, false, false);
}

union BFV { v16bf v; uint4 u[2]; };

__global__ __launch_bounds__(256) void gemm_ln_kernel(
    const float* __restrict__ x, const unsigned short* __restrict__ bw,
    const float* __restrict__ svec, const float* __restrict__ exit_w,
    const float* __restrict__ exit_b, const float* __restrict__ skip_b,
    const float* __restrict__ gamma, const float* __restrict__ beta,
    float* __restrict__ out) {
  __shared__ __align__(16) unsigned short Abuf[32 * LDA];
  __shared__ float sTile[32][NQ];
  __shared__ float psum[32][8];
  __shared__ float psq[32][8];
  __shared__ float muS[32];
  __shared__ float rsS[32];

  const int tid = threadIdx.x;
  const int lane = tid & 31;
  const int wv = tid >> 5;
  const int row0 = blockIdx.x * 32;

  {  // stage x tile (32 x 512) f32 -> bf16 LDS
    const int r = tid >> 3;
    const int k0 = (tid & 7) * 64;
    const float* xr = x + (size_t)(row0 + r) * IN_DIM + k0;
    unsigned short* dst = Abuf + r * LDA + k0;
#pragma unroll
    for (int j = 0; j < 64; j += 4) {
      float4 v = *(const float4*)(xr + j);
      dst[j + 0] = f2bf(v.x); dst[j + 1] = f2bf(v.y);
      dst[j + 2] = f2bf(v.z); dst[j + 3] = f2bf(v.w);
    }
  }
  if (tid < 32 * NQ)
    sTile[tid / NQ][tid % NQ] = svec[(size_t)(row0 + tid / NQ) * NQ + tid % NQ];
  __syncthreads();

  const int kh = lane >> 4;        // K-half select per 16-bit WMMA layout
  const int l16 = lane & 15;       // M (A) / N (B) index within tile
  const int ncol0 = wv * 64;

  v8f acc[2][4] = {};
  for (int kk = 0; kk < IN_DIM; kk += 32) {
    BFV af[2];
#pragma unroll
    for (int mt = 0; mt < 2; ++mt) {
      const unsigned short* ap = Abuf + (mt * 16 + l16) * LDA + kk + kh * 8;
      af[mt].u[0] = *(const uint4*)(ap);        // K[0..8)/K[8..16)
      af[mt].u[1] = *(const uint4*)(ap + 16);   // K[16..24)/K[24..32)
    }
#pragma unroll
    for (int nt = 0; nt < 4; ++nt) {
      const unsigned short* bp =
          bw + (size_t)(ncol0 + nt * 16 + l16) * IN_DIM + kk + kh * 8;
      BFV bf;
      bf.u[0] = *(const uint4*)(bp);
      bf.u[1] = *(const uint4*)(bp + 16);
      acc[0][nt] = wmma_bf16(af[0].v, bf.v, acc[0][nt]);
      acc[1][nt] = wmma_bf16(af[1].v, bf.v, acc[1][nt]);
    }
  }

  // epilogue params per N column
  float ewv[4][NQ], biasv[4], gmv[4], btv[4];
#pragma unroll
  for (int nt = 0; nt < 4; ++nt) {
    int n = ncol0 + nt * 16 + l16;
#pragma unroll
    for (int j = 0; j < NQ; ++j) ewv[nt][j] = exit_w[n * NQ + j];
    biasv[nt] = exit_b[n] + skip_b[n];
    gmv[nt] = gamma[n];
    btv[nt] = beta[n];
  }

  // add exit GEMM + biases; per-row partial sums for LayerNorm
#pragma unroll
  for (int mt = 0; mt < 2; ++mt) {
#pragma unroll
    for (int v = 0; v < 8; ++v) {
      const int m = mt * 16 + kh * 8 + v;
      float s0 = sTile[m][0], s1 = sTile[m][1], s2 = sTile[m][2];
      float s3 = sTile[m][3], s4 = sTile[m][4], s5 = sTile[m][5];
      float sm = 0.f, sq = 0.f;
#pragma unroll
      for (int nt = 0; nt < 4; ++nt) {
        float h = acc[mt][nt][v] + biasv[nt]
                + s0 * ewv[nt][0] + s1 * ewv[nt][1] + s2 * ewv[nt][2]
                + s3 * ewv[nt][3] + s4 * ewv[nt][4] + s5 * ewv[nt][5];
        acc[mt][nt][v] = h;
        sm += h;
        sq += h * h;
      }
#pragma unroll
      for (int msk = 8; msk >= 1; msk >>= 1) {   // reduce within 16-lane half
        sm += __shfl_xor(sm, msk, 32);
        sq += __shfl_xor(sq, msk, 32);
      }
      if (l16 == 0) { psum[m][wv] = sm; psq[m][wv] = sq; }
    }
  }
  __syncthreads();
  if (tid < 32) {
    float sm = 0.f, sq = 0.f;
#pragma unroll
    for (int w = 0; w < 8; ++w) { sm += psum[tid][w]; sq += psq[tid][w]; }
    float mu = sm * (1.f / 512.f);
    float var = sq * (1.f / 512.f) - mu * mu;
    muS[tid] = mu;
    rsS[tid] = rsqrtf(var + 1e-5f);
  }
  __syncthreads();

#pragma unroll
  for (int mt = 0; mt < 2; ++mt) {
#pragma unroll
    for (int v = 0; v < 8; ++v) {
      const int m = mt * 16 + kh * 8 + v;
      float mu = muS[m], rs = rsS[m];
      float* op = out + (size_t)(row0 + m) * OUT_DIM;
#pragma unroll
      for (int nt = 0; nt < 4; ++nt) {
        int n = ncol0 + nt * 16 + l16;
        op[n] = gmv[nt] * (acc[mt][nt][v] - mu) * rs + btv[nt];
      }
    }
  }
}

// ---------- launch ----------

extern "C" void kernel_launch(void* const* d_in, const int* in_sizes, int n_in,
                              void* d_out, int out_size, void* d_ws, size_t ws_size,
                              hipStream_t stream) {
  (void)in_sizes; (void)n_in; (void)out_size; (void)ws_size;
  const float* x       = (const float*)d_in[0];
  const float* entry_w = (const float*)d_in[1];
  const float* entry_b = (const float*)d_in[2];
  const float* exit_w  = (const float*)d_in[3];
  const float* exit_b  = (const float*)d_in[4];
  const float* skip_w  = (const float*)d_in[5];
  const float* skip_b  = (const float*)d_in[6];
  const float* vqc_w   = (const float*)d_in[7];
  const float* gamma   = (const float*)d_in[8];
  const float* beta    = (const float*)d_in[9];
  float* out = (float*)d_out;

  float* s_ws = (float*)d_ws;                                         // 32768*6 f32 = 768 KB
  unsigned short* bw = (unsigned short*)((char*)d_ws + (size_t)ROWS * NQ * 4);  // 512 KB bf16

  cvt_wb_kernel<<<(OUT_DIM * IN_DIM + 255) / 256, 256, 0, stream>>>(skip_w, bw, OUT_DIM * IN_DIM);
  vqc_kernel<<<ROWS / 8, 256, 0, stream>>>(x, entry_w, entry_b, vqc_w, s_ws);
  gemm_ln_kernel<<<ROWS / 32, 256, 0, stream>>>(x, bw, s_ws, exit_w, exit_b, skip_b,
                                                gamma, beta, out);
}
